// GRUDBackbone_58514634441385
// MI455X (gfx1250) — compile-verified, matrix operand order checked by reading, not run on hardware
//
#include <hip/hip_runtime.h>
#include <hip/hip_bf16.h>

// GRU-D backbone, CDNA5 (gfx1250, wave32, WMMA bf16).
// B=256, T=512, D=64, H=256.
//
// Pipeline:
//   k0: pack fp32 weights -> bf16 WMMA B-fragment layout (w_hh, w_ih, h_decay_w, x_decay_w)
//   k1: gamma_h = exp(-relu(td @ Whd^T + b))   (131072x64)@(64x256)  WMMA
//   k1: gamma_x = exp(-relu(td @ Wxd^T + b))   (131072x64)@(64x64)   WMMA
//   k2: x_last carry scan + x_hat + observed-mask (elementwise, parallel over (b,d))
//   k3: gi = [x_hat|m] @ Wih^T + b_ih          (131072x128)@(128x768) WMMA
//   k4: h recurrence: 16 WGs (16 batch rows each), 16 waves each, 512 steps;
//       h_tilde staged in LDS BOTH as fp32 (gate math) and pre-packed bf16
//       WMMA A-fragments (single ds_load_b128x2 per K-step); next-step
//       gamma_h/gi rows prefetched (global_prefetch_b8) under the WMMAs.

#define B_  256
#define T_  512
#define D_  64
#define H_  256
#define R_  (B_*T_)     // 131072 rows for batched GEMMs
#define G3H 768         // 3*H

typedef __attribute__((ext_vector_type(16))) __bf16 v16bf;
typedef __attribute__((ext_vector_type(8)))  __bf16 v8bf;
typedef __attribute__((ext_vector_type(8)))  float  v8f;

// CDNA5 16-bit WMMA fragment K-mapping (ISA 7.12.2):
// lanes 0-15 hold K = {0..7, 16..23}; lanes 16-31 hold K = {8..15, 24..31},
// 2 values packed per VGPR -> element i of v16bf:
__device__ __forceinline__ int kOfElem(int i, int hi) {
  return (i < 8 ? i : i + 8) + (hi ? 8 : 0);
}

__device__ __forceinline__ v8f wmma_bf16(v16bf a, v16bf b, v8f c) {
  return __builtin_amdgcn_wmma_f32_16x16x32_bf16(false, a, false, b, (short)0, c,
                                                 false, false);
}

// A-fragment (16x32, M x K) from fp32 row-major source, rows r0..r0+15.
__device__ __forceinline__ v16bf load_a_f32(const float* A, int lda, size_t r0, int k0, int lane) {
  const float* row = A + (r0 + (size_t)(lane & 15)) * lda + k0;
  int hi = lane >> 4;
  v16bf a;
#pragma unroll
  for (int i = 0; i < 16; ++i) a[i] = (__bf16)row[kOfElem(i, hi)];
  return a;
}

// A-fragment from bf16 row-major source.
__device__ __forceinline__ v16bf load_a_bf16(const __bf16* A, int lda, size_t r0, int k0, int lane) {
  const __bf16* row = A + (r0 + (size_t)(lane & 15)) * lda + k0;
  int hi = lane >> 4;
  v16bf a;
#pragma unroll
  for (int i = 0; i < 16; ++i) a[i] = row[kOfElem(i, hi)];
  return a;
}

__device__ __forceinline__ float fast_sigmoid(float x) {
  return 1.f / (1.f + __expf(-x));
}
__device__ __forceinline__ float fast_tanh(float x) {
  // tanh(x) = 1 - 2/(exp(2x)+1): stays on v_exp_f32 TRANS path (co-issues w/ WMMA)
  return 1.f - 2.f / (__expf(2.f * x) + 1.f);
}

// ---------------- k0: pack W (N rows x K cols, fp32) into bf16 B-fragments.
// packed[nTile][kTile][lane][i] = bf16( W[nTile*16 + lane%16][kTile*32 + kOfElem(i, lane>=16)] )
__global__ void pack_b_kernel(const float* __restrict__ W, __bf16* __restrict__ packed,
                              int N, int K) {
  int idx = blockIdx.x * blockDim.x + threadIdx.x;
  if (idx >= N * K) return;
  int i    = idx & 15;
  int lane = (idx >> 4) & 31;
  int lin  = idx >> 9;             // nTile*KT + kTile
  int KT   = K >> 5;
  int kt   = lin % KT, nt = lin / KT;
  int n = nt * 16 + (lane & 15);
  int k = kt * 32 + kOfElem(i, lane >> 4);
  packed[idx] = (__bf16)W[(size_t)n * K + k];
}

// ---------------- k1: gamma = exp(-relu(td @ W^T + b)), td is (R_, 64) fp32.
// One 16x16 output tile per wave; 2 K-iterations (K=64).
__global__ void decay_gemm_kernel(const float* __restrict__ td,
                                  const __bf16* __restrict__ packedW,
                                  const float* __restrict__ bias,
                                  __bf16* __restrict__ gamma, int N) {
  int lane = threadIdx.x & 31, wave = threadIdx.x >> 5;
  int tile = blockIdx.x * (blockDim.x >> 5) + wave;
  int nTiles = N >> 4;
  size_t r0 = (size_t)(tile / nTiles) * 16;
  int nt = tile % nTiles;
  const int KT = 2;
  v8f acc = {};
#pragma unroll
  for (int kt = 0; kt < KT; ++kt) {
    v16bf a = load_a_f32(td, D_, r0, kt * 32, lane);
    v16bf b = *(const v16bf*)(packedW + (size_t)((nt * KT + kt) * 32 + lane) * 16);
    acc = wmma_bf16(a, b, acc);
  }
  int n = nt * 16 + (lane & 15);
  float bn = bias[n];
  int hi = lane >> 4;
#pragma unroll
  for (int e = 0; e < 8; ++e) {
    size_t r = r0 + e + hi * 8;
    float v = acc[e] + bn;
    v = v > 0.f ? v : 0.f;
    gamma[r * N + n] = (__bf16)__expf(-v);
  }
}

// ---------------- k2: x_last carry scan (parallel over (b,d), sequential over t).
__global__ void xscan_kernel(const float* __restrict__ x, const int* __restrict__ lengths,
                             const float* __restrict__ missing, const float* __restrict__ means,
                             const __bf16* __restrict__ gamma_x,
                             __bf16* __restrict__ xhat, __bf16* __restrict__ mobs) {
  int tid = blockIdx.x * blockDim.x + threadIdx.x;   // 16384 threads
  int b = tid >> 6, d = tid & 63;
  float mean = means[d];
  float xl = mean;                                   // x_last0 = feature mean
  int len = lengths[b];
  size_t base = (size_t)b * T_ * D_ + d;
  for (int t = 0; t < T_; ++t) {
    size_t idx = base + (size_t)t * D_;
    float xv = x[idx];
    float m  = 1.f - missing[idx];                   // observed
    float g  = (float)gamma_x[idx];
    float xh = m * xv + (1.f - m) * (g * xl + (1.f - g) * mean);
    xhat[idx] = (__bf16)xh;
    mobs[idx] = (__bf16)m;
    float cand = m * xv + (1.f - m) * xl;
    if (t < len) xl = cand;                          // valid-gated carry
  }
}

// ---------------- k3: gi = [x_hat | m] @ Wih^T + b_ih  -> bf16 (R_, 768)
__global__ void gi_gemm_kernel(const __bf16* __restrict__ xhat, const __bf16* __restrict__ mobs,
                               const __bf16* __restrict__ packedWih,
                               const float* __restrict__ b_ih, __bf16* __restrict__ gi) {
  int lane = threadIdx.x & 31, wave = threadIdx.x >> 5;
  int tile = blockIdx.x * (blockDim.x >> 5) + wave;
  const int nTiles = G3H / 16, KT = 4;               // K = 128
  size_t r0 = (size_t)(tile / nTiles) * 16;
  int nt = tile % nTiles;
  v8f acc = {};
#pragma unroll
  for (int kt = 0; kt < KT; ++kt) {
    const __bf16* src = (kt < 2) ? xhat : mobs;
    v16bf a = load_a_bf16(src, D_, r0, (kt & 1) * 32, lane);
    v16bf b = *(const v16bf*)(packedWih + (size_t)((nt * KT + kt) * 32 + lane) * 16);
    acc = wmma_bf16(a, b, acc);
  }
  int n = nt * 16 + (lane & 15);
  float bn = b_ih[n];
  int hi = lane >> 4;
#pragma unroll
  for (int e = 0; e < 8; ++e) {
    size_t r = r0 + e + hi * 8;
    gi[r * G3H + n] = (__bf16)(acc[e] + bn);
  }
}

// ---------------- k4: sequential h recurrence. 16 WGs x 512 threads (16 waves).
__global__ void __launch_bounds__(512)
grud_recurrence_kernel(const __bf16* __restrict__ gamma_h, const __bf16* __restrict__ gi,
                       const __bf16* __restrict__ packedWhh, const float* __restrict__ b_hh,
                       const int* __restrict__ lengths, float* __restrict__ out) {
  __shared__ float h_s[16 * H_];                 // fp32 hidden state, 16 batch rows
  __shared__ float ht_s[16 * H_];                // fp32 h_tilde (for z*h_tilde gate term)
  __shared__ alignas(32) __bf16 afrag_s[8 * 32 * 16]; // pre-packed bf16 A-fragments [kt][lane][i]
  int b0 = blockIdx.x * 16;
  int tid = threadIdx.x;
  int lane = tid & 31, wave = tid >> 5;
  for (int i = tid; i < 16 * H_; i += 512) h_s[i] = 0.f;
  int nloc = lane & 15, hi = lane >> 4;
  int col = wave * 16 + nloc;                    // wave owns h-cols [16w,16w+16)
  float bh_r = b_hh[col], bh_z = b_hh[H_ + col], bh_n = b_hh[2 * H_ + col];
  int lenv[8];
#pragma unroll
  for (int e = 0; e < 8; ++e) lenv[e] = lengths[b0 + e + hi * 8];

  // phase-1 geometry: this thread owns rows m, k-block [k0,k0+8) of the 16x256 tile
  int m1  = tid >> 5;                            // 0..15
  int k0  = (tid & 31) * 8;                      // 0..248, step 8
  int kt1 = k0 >> 5;
  int kl  = k0 & 31;                             // in {0,8,16,24}
  int hi1 = (kl >> 3) & 1;
  int i0  = (kl & 16) ? 8 : 0;
  __bf16* fragDst = afrag_s + ((size_t)(kt1 * 32 + (m1 + 16 * hi1)) * 16 + i0);
  const __bf16* gRow = gamma_h + ((size_t)(b0 + m1) * T_) * H_ + k0;  // + t*H_ per step
  __syncthreads();

  for (int t = 0; t < T_; ++t) {
    // ---- phase 1: h_tilde = gamma_h[:, t, :] * h; write fp32 copy + bf16 A-fragment
    {
      const __bf16* gp = gRow + (size_t)t * H_;
      v8bf g8 = *(const v8bf*)gp;                // 16B vector load of gamma_h
      int idx0 = tid * 8;
      v8bf pk;
#pragma unroll
      for (int i = 0; i < 8; ++i) {
        float htv = (float)g8[i] * h_s[idx0 + i];
        ht_s[idx0 + i] = htv;
        pk[i] = (__bf16)htv;
      }
      *(v8bf*)fragDst = pk;                      // single 16B LDS store into fragment slot
      if (t + 1 < T_)                            // warm next step's gamma_h row
        __builtin_prefetch((const void*)(gp + H_), 0, 3);
    }
    __syncthreads();

    // warm next step's gi rows while WMMAs run (lanes 0-15 cover the 16 batch rows)
    if (t + 1 < T_) {
      size_t pb = ((size_t)(b0 + nloc) * T_ + (t + 1)) * G3H;
      __builtin_prefetch((const void*)(gi + pb + col), 0, 3);
      __builtin_prefetch((const void*)(gi + pb + H_ + col), 0, 3);
      __builtin_prefetch((const void*)(gi + pb + 2 * H_ + col), 0, 3);
    }

    // ---- phase 2: gh tiles {w, 16+w, 32+w} = (r,z,n) for cols [16w,16w+16), K=256
    v8f ar = {}, az = {}, an_ = {};
#pragma unroll
    for (int kt = 0; kt < 8; ++kt) {
      v16bf a = *(const v16bf*)(afrag_s + (size_t)(kt * 32 + lane) * 16); // 32B LDS load
      v16bf br = *(const v16bf*)(packedWhh + (size_t)(((wave)      * 8 + kt) * 32 + lane) * 16);
      v16bf bz = *(const v16bf*)(packedWhh + (size_t)(((16 + wave) * 8 + kt) * 32 + lane) * 16);
      v16bf bn = *(const v16bf*)(packedWhh + (size_t)(((32 + wave) * 8 + kt) * 32 + lane) * 16);
      ar  = wmma_bf16(a, br, ar);
      az  = wmma_bf16(a, bz, az);
      an_ = wmma_bf16(a, bn, an_);
    }

    // ---- fused GRU gates + h update + output (in-wave, cols owned exclusively)
#pragma unroll
    for (int e = 0; e < 8; ++e) {
      int m = e + hi * 8;
      int b = b0 + m;
      size_t gbase = ((size_t)b * T_ + t) * G3H;
      float gir = (float)gi[gbase + col];
      float giz = (float)gi[gbase + H_ + col];
      float gin = (float)gi[gbase + 2 * H_ + col];
      float r = fast_sigmoid(gir + ar[e] + bh_r);
      float z = fast_sigmoid(giz + az[e] + bh_z);
      float n = fast_tanh(gin + r * (an_[e] + bh_n));
      float htv = ht_s[m * H_ + col];
      float hc = (1.f - z) * n + z * htv;
      bool valid = t < lenv[e];
      float hold = h_s[m * H_ + col];
      h_s[m * H_ + col] = valid ? hc : hold;
      out[((size_t)b * T_ + t) * H_ + col] = valid ? hc : 0.f;
    }
    __syncthreads();
  }
}

extern "C" void kernel_launch(void* const* d_in, const int* in_sizes, int n_in,
                              void* d_out, int out_size, void* d_ws, size_t ws_size,
                              hipStream_t stream) {
  const float* x         = (const float*)d_in[0];
  const int*   lengths   = (const int*)  d_in[1];
  const float* missing   = (const float*)d_in[2];
  const float* time_dt   = (const float*)d_in[3];
  const float* means     = (const float*)d_in[4];
  const float* x_decay_w = (const float*)d_in[5];
  const float* x_decay_b = (const float*)d_in[6];
  const float* h_decay_w = (const float*)d_in[7];
  const float* h_decay_b = (const float*)d_in[8];
  const float* w_ih      = (const float*)d_in[9];
  const float* w_hh      = (const float*)d_in[10];
  const float* b_ih      = (const float*)d_in[11];
  const float* b_hh      = (const float*)d_in[12];
  float* out = (float*)d_out;

  // workspace layout (bytes, 256-aligned)
  char* ws = (char*)d_ws;
  size_t off = 0;
  auto take = [&](size_t bytes) { char* p = ws + off; off = (off + bytes + 255) & ~(size_t)255; return p; };
  __bf16* pWhh   = (__bf16*)take((size_t)G3H * H_  * 2);   // 48x8 fragments
  __bf16* pWih   = (__bf16*)take((size_t)G3H * 128 * 2);   // 48x4 fragments
  __bf16* pWhd   = (__bf16*)take((size_t)H_  * D_  * 2);   // 16x2 fragments
  __bf16* pWxd   = (__bf16*)take((size_t)D_  * D_  * 2);   // 4x2 fragments
  __bf16* gammaH = (__bf16*)take((size_t)R_ * H_ * 2);
  __bf16* gammaX = (__bf16*)take((size_t)R_ * D_ * 2);
  __bf16* xhat   = (__bf16*)take((size_t)R_ * D_ * 2);
  __bf16* mobs   = (__bf16*)take((size_t)R_ * D_ * 2);
  __bf16* gi     = (__bf16*)take((size_t)R_ * G3H * 2);
  (void)ws_size; (void)in_sizes; (void)n_in; (void)out_size;

  // k0: pack weights into bf16 B-fragment layout
  pack_b_kernel<<<(G3H * H_  + 255) / 256, 256, 0, stream>>>(w_hh,      pWhh, G3H, H_);
  pack_b_kernel<<<(G3H * 128 + 255) / 256, 256, 0, stream>>>(w_ih,      pWih, G3H, 128);
  pack_b_kernel<<<(H_  * D_  + 255) / 256, 256, 0, stream>>>(h_decay_w, pWhd, H_,  D_);
  pack_b_kernel<<<(D_  * D_  + 255) / 256, 256, 0, stream>>>(x_decay_w, pWxd, D_,  D_);

  // k1: decay GEMMs (8 waves/block, 1 tile/wave)
  decay_gemm_kernel<<<(R_ / 16) * (H_ / 16) / 8, 256, 0, stream>>>(time_dt, pWhd, h_decay_b, gammaH, H_);
  decay_gemm_kernel<<<(R_ / 16) * (D_ / 16) / 8, 256, 0, stream>>>(time_dt, pWxd, x_decay_b, gammaX, D_);

  // k2: x_last scan
  xscan_kernel<<<(B_ * D_) / 256, 256, 0, stream>>>(x, lengths, missing, means, gammaX, xhat, mobs);

  // k3: gi GEMM
  gi_gemm_kernel<<<(R_ / 16) * (G3H / 16) / 8, 256, 0, stream>>>(xhat, mobs, pWih, b_ih, gi);

  // k4: recurrence — 16 WGs, one per 16-row batch tile, independent over B
  grud_recurrence_kernel<<<B_ / 16, 512, 0, stream>>>(gammaH, gi, pWhh, b_hh, lengths, out);
}
